// BiaffineAttn_29695403885100
// MI455X (gfx1250) — compile-verified, matrix operand order checked by reading, not run on hardware
//
#include <hip/hip_runtime.h>
#include <hip/hip_bf16.h>

#define BB 8
#define SS 2048
#define DD 1024
#define EE 512

typedef __attribute__((ext_vector_type(16))) __bf16 v16bf;
typedef __attribute__((ext_vector_type(8)))  __bf16 v8bf;
typedef __attribute__((ext_vector_type(8)))  float  v8f;
typedef __attribute__((ext_vector_type(4)))  unsigned int u32x4;
typedef __attribute__((ext_vector_type(8)))  int i32x8;
typedef __attribute__((ext_vector_type(4)))  int i32x4;

union BF16x16 { v16bf v; v8bf h[2]; };

__device__ __forceinline__ unsigned short f2bf(float f) {
    unsigned int u = __float_as_uint(f);
    u += 0x7FFFu + ((u >> 16) & 1u);           // round-to-nearest-even
    return (unsigned short)(u >> 16);
}

// Load 16 bf16 elements as two 16B chunks: {k..k+7} and {k+16..k+23}
// (caller passes p already offset by k0 + 8*half, matching the ISA 16-bit
//  A/B VGPR layout for wave32 WMMA)
__device__ __forceinline__ v16bf load_bf16x16(const unsigned short* p) {
    BF16x16 u;
    u.h[0] = *reinterpret_cast<const v8bf*>(p);
    u.h[1] = *reinterpret_cast<const v8bf*>(p + 16);
    return u.v;
}

__device__ __forceinline__ v8f wmma_bf16(v16bf a, v16bf b, v8f c) {
    return __builtin_amdgcn_wmma_f32_16x16x32_bf16(
        /*neg_a=*/false, a, /*neg_b=*/false, b,
        /*c_mod=*/(short)0, c, /*reuse_a=*/false, /*reuse_b=*/false);
}

// ---- Tensor Data Mover: 2D tile (bf16 elements) global -> LDS ----
// tensor: rows with leading-dim stride_elems (elements); tile: tile_rows x tile_k,
// stored contiguously in LDS as [row][tile_k].
__device__ __forceinline__ void tdm_load_2d(unsigned lds_addr, const void* gaddr,
                                            unsigned tensor_k, unsigned tensor_rows,
                                            unsigned tile_k, unsigned tile_rows,
                                            unsigned stride_elems) {
    unsigned long long ga = (unsigned long long)(uintptr_t)gaddr;
    u32x4 g0;
    g0[0] = 1u;                                        // count=1 valid user D#
    g0[1] = lds_addr;                                  // LDS byte address
    g0[2] = (unsigned)(ga & 0xFFFFFFFFu);              // global_addr[31:0]
    g0[3] = (unsigned)((ga >> 32) & 0x01FFFFFFu) | (2u << 30);  // addr[56:32] | type=2
    i32x8 g1;
    g1[0] = (int)(1u << 16);                           // data_size = 2 bytes
    g1[1] = (int)((tensor_k & 0xFFFFu) << 16);         // tensor_dim0[15:0]
    g1[2] = (int)(((tensor_k >> 16) & 0xFFFFu) |
                  ((tensor_rows & 0xFFFFu) << 16));    // dim0[31:16] | dim1[15:0]
    g1[3] = (int)(((tensor_rows >> 16) & 0xFFFFu) |
                  ((tile_k & 0xFFFFu) << 16));         // dim1[31:16] | tile_dim0
    g1[4] = (int)(tile_rows & 0xFFFFu);                // tile_dim1, tile_dim2=0
    g1[5] = (int)stride_elems;                         // tensor_dim0_stride[31:0]
    g1[6] = 0;                                         // stride0[47:32]=0, stride1 lo=0
    g1[7] = 0;
    i32x4 z4 = {0, 0, 0, 0};
    i32x8 z8 = {0, 0, 0, 0, 0, 0, 0, 0};
    __builtin_amdgcn_tensor_load_to_lds(g0, g1, z4, z4, z8, 0);
}

__device__ __forceinline__ unsigned lds_off(const void* p) {
    return (unsigned)(uintptr_t)p;   // flat aperture: addr[31:0] == LDS offset
}

// ---------------- prep kernels ----------------

__global__ void cvt_bf16_k(const float* __restrict__ in,
                           unsigned short* __restrict__ out, int n) {
    int i = blockIdx.x * 256 + threadIdx.x;
    if (i < n) out[i] = f2bf(in[i]);
}

// LDS-tiled transpose: out[b][c][r] = bf16(in[b][r][c]); both sides coalesced.
// grid = (C/32, R/32, Bn), block = 256.
__global__ void transpose_tiled_k(const float* __restrict__ in,
                                  unsigned short* __restrict__ out,
                                  int R, int C) {
    __shared__ float tile[32][33];
    const int b  = blockIdx.z;
    const int c0 = blockIdx.x * 32;
    const int r0 = blockIdx.y * 32;
    const int tx = threadIdx.x & 31;
    const int ty = threadIdx.x >> 5;           // 0..7
    const float* src = in + ((size_t)b * R + r0) * C + c0;
#pragma unroll
    for (int i = 0; i < 4; ++i)
        tile[ty + 8 * i][tx] = src[(size_t)(ty + 8 * i) * C + tx];
    __syncthreads();
    unsigned short* dst = out + ((size_t)b * C + c0) * R + r0;
#pragma unroll
    for (int i = 0; i < 4; ++i)
        dst[(size_t)(ty + 8 * i) * R + tx] = f2bf(tile[tx][ty + 8 * i]);
}

// out[row] = dot(x[row, :], v)   (one wave32 per row)
__global__ void rowdot_k(const float* __restrict__ x, const float* __restrict__ v,
                         float* __restrict__ out, int D) {
    int row  = blockIdx.x * 8 + (threadIdx.x >> 5);
    int lane = threadIdx.x & 31;
    const float* p = x + (size_t)row * D;
    float s = 0.f;
    for (int i = lane; i < D; i += 32) s += p[i] * v[i];
#pragma unroll
    for (int off = 16; off > 0; off >>= 1) s += __shfl_xor(s, off, 32);
    if (lane == 0) out[row] = s;
}

// ---------------- TDM-staged bf16 WMMA GEMM ----------------
// C[M,N] = A[M,K] (row-major bf16) * Bc[N,K] (B column-major: row n == column n).
// Block: 16 rows x 512 cols; 8 waves, each wave 16x64 (4 accumulator tiles).
// A/B panels are DMA'd into LDS by the Tensor Data Mover, double-buffered in
// KC=64 K-chunks; sync = s_wait_tensorcnt (issuing wave) + workgroup barriers.
#define KC 64
__global__ void gemm_bf16_wmma(const unsigned short* __restrict__ A,
                               const unsigned short* __restrict__ Bc,
                               const float* __restrict__ bias,
                               float* __restrict__ outF,
                               unsigned short* __restrict__ outB,
                               int N, int K, int relu) {
    extern __shared__ char smem[];
    unsigned short* sA = (unsigned short*)smem;                 // 2 * 16  * KC
    unsigned short* sB = (unsigned short*)(smem + 2 * 16 * KC * 2); // 2 * 512 * KC

    const int tid  = threadIdx.x;
    const int w    = tid >> 5;
    const int lane = tid & 31;
    const int nn   = lane & 15;
    const int hh   = lane >> 4;
    const int row0 = blockIdx.x * 16;
    const int colB = blockIdx.y * 512;

    const unsigned short* Ablk = A  + (size_t)row0 * K;
    const unsigned short* Bblk = Bc + (size_t)colB * K;

    if (w == 0) {   // prefetch chunk 0
        tdm_load_2d(lds_off(sA), Ablk, K, 16,  KC, 16,  K);
        tdm_load_2d(lds_off(sB), Bblk, K, 512, KC, 512, K);
    }

    const v8f zero = {0.f, 0.f, 0.f, 0.f, 0.f, 0.f, 0.f, 0.f};
    v8f acc[4];
#pragma unroll
    for (int i = 0; i < 4; ++i) acc[i] = zero;

    const int nchunk = K / KC;
    for (int c = 0; c < nchunk; ++c) {
        if (w == 0) __builtin_amdgcn_s_wait_tensorcnt(0);  // chunk c resident
        __syncthreads();
        if (w == 0 && c + 1 < nchunk) {                    // start DMA of chunk c+1
            int k0 = (c + 1) * KC;
            int nb = (c + 1) & 1;
            tdm_load_2d(lds_off(sA) + nb * 16 * KC * 2,  Ablk + k0, K, 16,  KC, 16,  K);
            tdm_load_2d(lds_off(sB) + nb * 512 * KC * 2, Bblk + k0, K, 512, KC, 512, K);
        }
        const unsigned short* aB = sA + (c & 1) * 16 * KC;
        const unsigned short* bB = sB + (c & 1) * 512 * KC;
#pragma unroll
        for (int kk = 0; kk < KC; kk += 32) {
            v16bf a = load_bf16x16(aB + nn * KC + kk + hh * 8);
#pragma unroll
            for (int nt = 0; nt < 4; ++nt) {
                const unsigned short* bp = bB + (w * 64 + nt * 16 + nn) * KC + kk + hh * 8;
                acc[nt] = wmma_bf16(a, load_bf16x16(bp), acc[nt]);
            }
        }
        __syncthreads();   // all waves done with buffer (c&1) before it is refilled
    }

#pragma unroll
    for (int nt = 0; nt < 4; ++nt) {
        int col = colB + w * 64 + nt * 16 + nn;
        float bv = bias ? bias[col] : 0.f;
#pragma unroll
        for (int r = 0; r < 8; ++r) {
            int m = r + 8 * hh;
            float v = acc[nt][r] + bv;
            if (relu) v = fmaxf(v, 0.f);
            size_t idx = (size_t)(row0 + m) * N + col;
            if (outB) outB[idx] = f2bf(v);
            else      outF[idx] = v;
        }
    }
}

// ---------------- fused attention: QK^T + bias + softmax + PV ----------------
// One block per (b, 16-row query tile). 256 threads = 8 waves.
// LDS: 16x2048 f32 scores (128KB) + 16x2048 bf16 P (64KB) + 16x1024 bf16 y-tile
// (32KB, TDM-loaded once) = 224KB of the 320KB WGP LDS.
__global__ void attn_fused(const unsigned short* __restrict__ yb,    // [B,S,D] bf16
                           const unsigned short* __restrict__ x1b,   // [B,S,D] bf16
                           const unsigned short* __restrict__ x1tb,  // [B,D,S] bf16
                           const float* __restrict__ keybias,        // [B,S]
                           unsigned short* __restrict__ Ob) {        // [B,S,D] bf16
    extern __shared__ char smem[];
    float* sS          = (float*)smem;                                      // [16][SS]
    unsigned short* sP = (unsigned short*)(smem + 16 * SS * sizeof(float)); // [16][SS]
    unsigned short* sY = sP + 16 * SS;                                      // [16][DD]

    const int b    = blockIdx.x >> 7;          // / (SS/16)
    const int s0   = (blockIdx.x & 127) << 4;
    const int tid  = threadIdx.x;
    const int w    = tid >> 5;
    const int lane = tid & 31;
    const int nn   = lane & 15;
    const int hh   = lane >> 4;

    // TDM: stage the whole 16x1024 y A-tile into LDS once.
    if (w == 0) {
        tdm_load_2d(lds_off(sY), yb + (size_t)(b * SS + s0) * DD, DD, 16, DD, 16, DD);
        __builtin_amdgcn_s_wait_tensorcnt(0);
    }
    __syncthreads();

    const v8f zero = {0.f, 0.f, 0.f, 0.f, 0.f, 0.f, 0.f, 0.f};

    // ---- QK: this wave produces scores[:, w*256 .. w*256+255] ----
    const unsigned short* yrow = sY + nn * DD;
    const int tcol0 = w * 256;
    v8f acc[16];
#pragma unroll
    for (int i = 0; i < 16; ++i) acc[i] = zero;

    for (int k0 = 0; k0 < DD; k0 += 32) {
        v16bf a = load_bf16x16(yrow + k0 + hh * 8);  // A reused across 16 N-tiles
#pragma unroll
        for (int nt = 0; nt < 16; ++nt) {
            const unsigned short* bp =
                x1b + (size_t)(b * SS + tcol0 + nt * 16 + nn) * DD + k0 + hh * 8;
            acc[nt] = wmma_bf16(a, load_bf16x16(bp), acc[nt]);
        }
    }

    // scores + key bias -> LDS
#pragma unroll
    for (int nt = 0; nt < 16; ++nt) {
        int col = tcol0 + nt * 16 + nn;
        float kb = keybias[b * SS + col];
#pragma unroll
        for (int r = 0; r < 8; ++r) {
            int m = r + 8 * hh;
            sS[m * SS + col] = acc[nt][r] + kb;
        }
    }
    __syncthreads();

    // ---- softmax: wave w handles rows 2w and 2w+1 ----
    for (int rr = 0; rr < 2; ++rr) {
        int m = 2 * w + rr;
        float mx = -3.402823466e38f;
        for (int t = lane; t < SS; t += 32) mx = fmaxf(mx, sS[m * SS + t]);
#pragma unroll
        for (int off = 16; off > 0; off >>= 1) mx = fmaxf(mx, __shfl_xor(mx, off, 32));
        float sum = 0.f;
        for (int t = lane; t < SS; t += 32) {
            float e = __expf(sS[m * SS + t] - mx);
            sS[m * SS + t] = e;
            sum += e;
        }
#pragma unroll
        for (int off = 16; off > 0; off >>= 1) sum += __shfl_xor(sum, off, 32);
        float inv = 1.0f / sum;
        for (int t = lane; t < SS; t += 32)
            sP[m * SS + t] = f2bf(sS[m * SS + t] * inv);
    }
    __syncthreads();

    // ---- PV: wave w owns output d-columns [w*128, w*128+128) ----
    v8f oacc[8];
#pragma unroll
    for (int i = 0; i < 8; ++i) oacc[i] = zero;

    const unsigned short* prow = sP + nn * SS;   // A rows of P tile from LDS
    for (int k0 = 0; k0 < SS; k0 += 32) {
        v16bf a = load_bf16x16(prow + k0 + hh * 8);
#pragma unroll
        for (int nt = 0; nt < 8; ++nt) {
            int dc = w * 128 + nt * 16 + nn;
            const unsigned short* bp =
                x1tb + (size_t)(b * DD + dc) * SS + k0 + hh * 8;
            oacc[nt] = wmma_bf16(a, load_bf16x16(bp), oacc[nt]);
        }
    }

#pragma unroll
    for (int nt = 0; nt < 8; ++nt) {
        int dc = w * 128 + nt * 16 + nn;
#pragma unroll
        for (int r = 0; r < 8; ++r) {
            int m = r + 8 * hh;
            Ob[(size_t)(b * SS + s0 + m) * DD + dc] = f2bf(oacc[nt][r]);
        }
    }
}

// ---------------- host ----------------

extern "C" void kernel_launch(void* const* d_in, const int* in_sizes, int n_in,
                              void* d_out, int out_size, void* d_ws, size_t ws_size,
                              hipStream_t stream) {
    const float* x1   = (const float*)d_in[0];
    const float* x2   = (const float*)d_in[1];
    const float* U    = (const float*)d_in[2];
    const float* bias = (const float*)d_in[3];
    const float* fcw  = (const float*)d_in[4];
    const float* fcb  = (const float*)d_in[5];
    float* out = (float*)d_out;

    char* ws = (char*)d_ws;
    const size_t nBSD = (size_t)BB * SS * DD;
    unsigned short* x1b  = (unsigned short*)ws; ws += nBSD * 2;
    unsigned short* x2b  = (unsigned short*)ws; ws += nBSD * 2;
    unsigned short* x1tb = (unsigned short*)ws; ws += nBSD * 2;
    unsigned short* Ucb  = (unsigned short*)ws; ws += (size_t)DD * DD * 2;
    unsigned short* fcwb = (unsigned short*)ws; ws += (size_t)EE * DD * 2;
    float* keybias       = (float*)ws;          ws += (size_t)BB * SS * 4;
    unsigned short* yb   = (unsigned short*)ws; ws += nBSD * 2;
    unsigned short* Ob   = (unsigned short*)ws; ws += nBSD * 2;

    const int n1 = (int)nBSD;   // 16,777,216

    cvt_bf16_k<<<n1 / 256, 256, 0, stream>>>(x1, x1b, n1);
    cvt_bf16_k<<<n1 / 256, 256, 0, stream>>>(x2, x2b, n1);
    cvt_bf16_k<<<(EE * DD) / 256, 256, 0, stream>>>(fcw, fcwb, EE * DD);

    // coalesced LDS-tiled transposes (bf16 output)
    dim3 gt1(DD / 32, SS / 32, BB);
    transpose_tiled_k<<<gt1, 256, 0, stream>>>(x1, x1tb, SS, DD);
    dim3 gtu(DD / 32, DD / 32, 1);
    transpose_tiled_k<<<gtu, 256, 0, stream>>>(U, Ucb, DD, DD);

    rowdot_k<<<(BB * SS) / 8, 256, 0, stream>>>(x1, bias, keybias, DD);

    // y = x2 @ U  -> bf16 [B,S,D]   (TDM-staged GEMM)
    size_t gemm_shmem = (size_t)2 * 16 * KC * 2 + (size_t)2 * 512 * KC * 2; // 132KB
    dim3 gy(BB * SS / 16, DD / 512);
    gemm_bf16_wmma<<<gy, 256, gemm_shmem, stream>>>(x2b, Ucb, nullptr, nullptr, yb,
                                                    DD, DD, 0);

    // fused attention: scores+bias+softmax+PV, 224KB LDS
    size_t shmem = (size_t)16 * SS * sizeof(float)
                 + (size_t)16 * SS * sizeof(unsigned short)
                 + (size_t)16 * DD * sizeof(unsigned short);
    attn_fused<<<BB * (SS / 16), 256, shmem, stream>>>(yb, x1b, x1tb, keybias, Ob);

    // out = relu(O @ fc_w^T + fc_b)  (fc_w is already [E][D] = B column-major)
    dim3 gf(BB * SS / 16, EE / 512);
    gemm_bf16_wmma<<<gf, 256, gemm_shmem, stream>>>(Ob, fcwb, fcb, out, nullptr,
                                                    EE, DD, 1);
}
